// TransformerEncoder_66984309948642
// MI455X (gfx1250) — matrix operand order repaired and checked by reading, not compiled, not run on hardware
//
#include <hip/hip_runtime.h>

typedef __attribute__((ext_vector_type(16))) _Float16 v16h;
typedef __attribute__((ext_vector_type(8)))  _Float16 v8h;
typedef __attribute__((ext_vector_type(8)))  float    v8f;

#define NROW   65536L   // B*S
#define DMODEL 256
#define NHEAD  8
#define DHEAD  32
#define FFDIM  1024
#define NLAYER 4
#define SEQ    4096
#define BATCH  16

// ---------------------------------------------------------------------------
// WMMA helpers (CDNA5 v_wmma_f32_16x16x32_f16)
// ---------------------------------------------------------------------------
__device__ __forceinline__ v8f wmma_f16(v16h a, v16h b, v8f c) {
  // (neg_a, A, neg_b, B, c_mod, C, reuse_a, reuse_b)
  return __builtin_amdgcn_wmma_f32_16x16x32_f16(false, a, false, b, (short)0, c,
                                                false, false);
}

// A fragment, 16x32 f16, A row-major with leading dim ld (elements).
// Layout: lanes 0-15: M=lane, elements hold K=kb..kb+15 contiguous
// where kb = (lane>=16) ? 16 : 0.
__device__ __forceinline__ v16h load_a_frag(const _Float16* A, long ld, int m0,
                                            int k0, int lane) {
  int m  = m0 + (lane & 15);
  int kb = k0 + ((lane >> 4) << 4);
  const _Float16* p = A + (long)m * ld + kb;
  v8h lo = *(const v8h*)(p);
  v8h hi = *(const v8h*)(p + 8);
  v16h r;
#pragma unroll
  for (int i = 0; i < 8; ++i) { r[i] = lo[i]; r[8 + i] = hi[i]; }
  return r;
}

// B fragment, 32x16 f16, stored transposed (Bt[N][K]) so each lane reads 16
// contiguous halves. lanes 0-15: N=lane, K=0..15; lanes 16-31: same N, K=16..31.
__device__ __forceinline__ v16h load_b_frag(const _Float16* Bt, long ld, int n0,
                                            int k0, int lane) {
  int n  = n0 + (lane & 15);
  int kb = k0 + ((lane >> 4) << 4);
  const _Float16* p = Bt + (long)n * ld + kb;
  v8h lo = *(const v8h*)(p);
  v8h hi = *(const v8h*)(p + 8);
  v16h r;
#pragma unroll
  for (int i = 0; i < 8; ++i) { r[i] = lo[i]; r[8 + i] = hi[i]; }
  return r;
}

// Column-major A fragment: A[m][k] lives at base[k*256 + m] (for ctx = k^T v).
__device__ __forceinline__ v16h load_a_colmajor(const _Float16* base, int m0,
                                                int k0, int lane) {
  int m  = m0 + (lane & 15);
  int kb = k0 + ((lane >> 4) << 4);
  v16h r;
#pragma unroll
  for (int i = 0; i < 8; ++i) {
    r[i]     = base[(long)(kb + i) * DMODEL + m];
    r[8 + i] = base[(long)(kb + 8 + i) * DMODEL + m];
  }
  return r;
}

// Row-major-B fragment: B[k][n] lives at base[k*256 + n] (v matrix, strided).
__device__ __forceinline__ v16h load_b_rowmajor(const _Float16* base, int n0,
                                                int k0, int lane) {
  int n  = n0 + (lane & 15);
  int kb = k0 + ((lane >> 4) << 4);
  v16h r;
#pragma unroll
  for (int i = 0; i < 16; ++i) r[i] = base[(long)(kb + i) * DMODEL + n];
  return r;
}

// ---------------------------------------------------------------------------
// Positional encoding add:  x = x_in + PE
// ---------------------------------------------------------------------------
__global__ void te_pe_add(const float* __restrict__ xin, float* __restrict__ xo) {
  long idx = (long)blockIdx.x * 256 + threadIdx.x;
  int d = (int)(idx & (DMODEL - 1));
  int s = (int)((idx >> 8) & (SEQ - 1));
  int i2 = d & ~1;
  float div = expf(-(float)i2 * 0.03597789207803197f);  // ln(10000)/256
  float arg = (float)s * div;
  float pe = (d & 1) ? cosf(arg) : sinf(arg);
  xo[idx] = xin[idx] + pe;
}

// ---------------------------------------------------------------------------
// LayerNorm (one wave per row of 256) -> f16
// ---------------------------------------------------------------------------
__global__ void te_layernorm(const float* __restrict__ x,
                             const float* __restrict__ g,
                             const float* __restrict__ b,
                             _Float16* __restrict__ h16) {
  int lane = threadIdx.x & 31;
  int wv   = threadIdx.x >> 5;
  long row = (long)blockIdx.x * 8 + wv;
  const float* xr = x + row * DMODEL + lane * 8;
  float v[8]; float s = 0.f, sq = 0.f;
#pragma unroll
  for (int i = 0; i < 8; ++i) { v[i] = xr[i]; s += v[i]; sq += v[i] * v[i]; }
#pragma unroll
  for (int m = 16; m > 0; m >>= 1) {
    s  += __shfl_xor(s,  m, 32);
    sq += __shfl_xor(sq, m, 32);
  }
  float mu  = s * (1.f / 256.f);
  float var = sq * (1.f / 256.f) - mu * mu;
  float rs  = rsqrtf(var + 1e-5f);
  const float* gp = g + lane * 8;
  const float* bp = b + lane * 8;
  _Float16* hr = h16 + row * DMODEL + lane * 8;
#pragma unroll
  for (int i = 0; i < 8; ++i)
    hr[i] = (_Float16)((v[i] - mu) * rs * gp[i] + bp[i]);
}

// ---------------------------------------------------------------------------
// Generic WMMA GEMM: out[M,N] = A[M,K](f16 row-major) x Wt[N,K](f16, B^T).
// Block = 8 waves = 128 rows x 64 cols of output. The 64xKC weight panel is
// staged into LDS with CDNA5 async global->LDS copies (ASYNCcnt), then all 8
// waves read B fragments from LDS (ds_load_b128), A fragments from global.
// ---------------------------------------------------------------------------
#define EP_F32        0
#define EP_F16        1
#define EP_BIAS_RES   2
#define EP_BIAS_GELU  3

#define KC 256  // K-panel staged in LDS: 64 * 256 * 2B = 32 KB

template <int EP, int K>
__global__ void te_gemm(const _Float16* __restrict__ A,
                        const _Float16* __restrict__ Wt,
                        const float* __restrict__ bias,
                        const float* __restrict__ resid,
                        void* __restrict__ out, int M, int N) {
  __shared__ _Float16 bsh[64 * KC];
  int lane = threadIdx.x & 31;
  int wave = threadIdx.x >> 5;
  int nb = N >> 6;
  int bn = (int)(blockIdx.x % nb);
  int bm = (int)(blockIdx.x / nb);
  int n0 = bn << 6;
  int m0 = (bm * 8 + wave) << 4;

  int nn = lane & 15;           // B-fragment column within 16-tile
  int kbb = (lane >> 4) << 4;   // B-fragment K sub-offset (0 or 16)

  v8f c0 = {}, c1 = {}, c2 = {}, c3 = {};

  for (int kc = 0; kc < K; kc += KC) {
    // ---- async-stage Wt[n0..n0+63][kc..kc+KC) into LDS (ASYNCcnt path) ----
#pragma unroll 1
    for (int t = threadIdx.x; t < (64 * KC) / 8; t += 256) {
      int n = t >> 5;               // row within panel
      int c = t & 31;               // 16B chunk within row
      unsigned lds_off = (unsigned)(unsigned long long)&bsh[n * KC + c * 8];
      unsigned long long ga =
          (unsigned long long)(Wt + (long)(n0 + n) * K + kc + c * 8);
      asm volatile("global_load_async_to_lds_b128 %0, %1, off"
                   :: "v"(lds_off), "v"(ga) : "memory");
    }
    asm volatile("s_wait_asynccnt 0x0" ::: "memory");
    __syncthreads();

    // ---- MACs over this K panel ----
    for (int k0 = 0; k0 < KC; k0 += 32) {
      if (k0 + 32 < KC)  // global_prefetch_b8 of next A chunk
        __builtin_prefetch(A + (long)m0 * K + kc + k0 + 32, 0, 0);
      v16h a = load_a_frag(A, K, m0, kc + k0, lane);
      v16h b0, b1, b2, b3;
#pragma unroll
      for (int i = 0; i < 8; ++i) {
        const _Float16* p0 = &bsh[(nn)      * KC + k0 + kbb];
        const _Float16* p1 = &bsh[(nn + 16) * KC + k0 + kbb];
        const _Float16* p2 = &bsh[(nn + 32) * KC + k0 + kbb];
        const _Float16* p3 = &bsh[(nn + 48) * KC + k0 + kbb];
        b0[i] = p0[i]; b0[8 + i] = p0[8 + i];
        b1[i] = p1[i]; b1[8 + i] = p1[8 + i];
        b2[i] = p2[i]; b2[8 + i] = p2[8 + i];
        b3[i] = p3[i]; b3[8 + i] = p3[8 + i];
      }
      c0 = wmma_f16(a, b0, c0);
      c1 = wmma_f16(a, b1, c1);
      c2 = wmma_f16(a, b2, c2);
      c3 = wmma_f16(a, b3, c3);
    }
    __syncthreads();  // waves done reading before next panel restage
  }

  int rb = (lane >> 4) << 3;
  int nc = lane & 15;
#pragma unroll
  for (int r = 0; r < 8; ++r) {
    long row = m0 + rb + r;
    float vals[4] = {c0[r], c1[r], c2[r], c3[r]};
#pragma unroll
    for (int j = 0; j < 4; ++j) {
      int col = n0 + j * 16 + nc;
      long idx = row * N + col;
      if (EP == EP_F32) {
        ((float*)out)[idx] = vals[j];
      } else if (EP == EP_F16) {
        ((_Float16*)out)[idx] = (_Float16)vals[j];
      } else if (EP == EP_BIAS_RES) {
        ((float*)out)[idx] = vals[j] + bias[col] + resid[idx];
      } else {  // bias + exact GELU -> f16
        float t = vals[j] + bias[col];
        t = 0.5f * t * (1.f + erff(t * 0.70710678118654752f));
        ((_Float16*)out)[idx] = (_Float16)t;
      }
    }
  }
}

// ---------------------------------------------------------------------------
// q feature softmax (over DH=32 within head) * scale -> f16.  One wave per
// (row, head) group of 32.
// ---------------------------------------------------------------------------
__global__ void te_qsoftmax(const float* __restrict__ q32,
                            _Float16* __restrict__ q16) {
  int lane = threadIdx.x & 31;
  int wv   = threadIdx.x >> 5;
  long w   = (long)blockIdx.x * 8 + wv;
  long row = w >> 3;
  int  h   = (int)(w & 7);
  long idx = row * DMODEL + h * DHEAD + lane;
  float t = q32[idx];
  float mx = t;
#pragma unroll
  for (int m = 16; m > 0; m >>= 1) mx = fmaxf(mx, __shfl_xor(mx, m, 32));
  float e = expf(t - mx);
  float ss = e;
#pragma unroll
  for (int m = 16; m > 0; m >>= 1) ss += __shfl_xor(ss, m, 32);
  q16[idx] = (_Float16)(e / ss * 0.17677669529663689f);  // DH^-0.5
}

// ---------------------------------------------------------------------------
// k sequence softmax (over S=4096 per (batch, column)) -> f16.
// One block per (b, c); 256 threads x 16 rows each.
// ---------------------------------------------------------------------------
__global__ void te_ksoftmax(const float* __restrict__ k32,
                            _Float16* __restrict__ k16) {
  __shared__ float red[256];
  int tid = threadIdx.x;
  int b = blockIdx.x >> 8;
  int c = blockIdx.x & 255;
  long base = (long)b * SEQ * DMODEL + c;
  float v[16];
  float lmax = -3.4e38f;
#pragma unroll
  for (int j = 0; j < 16; ++j) {
    v[j] = k32[base + (long)(tid + j * 256) * DMODEL];
    lmax = fmaxf(lmax, v[j]);
  }
  red[tid] = lmax; __syncthreads();
  for (int s = 128; s > 0; s >>= 1) {
    if (tid < s) red[tid] = fmaxf(red[tid], red[tid + s]);
    __syncthreads();
  }
  float mx = red[0]; __syncthreads();
  float ls = 0.f;
#pragma unroll
  for (int j = 0; j < 16; ++j) { v[j] = expf(v[j] - mx); ls += v[j]; }
  red[tid] = ls; __syncthreads();
  for (int s = 128; s > 0; s >>= 1) {
    if (tid < s) red[tid] += red[tid + s];
    __syncthreads();
  }
  float inv = 1.f / red[0];
#pragma unroll
  for (int j = 0; j < 16; ++j)
    k16[base + (long)(tid + j * 256) * DMODEL] = (_Float16)(v[j] * inv);
}

// ---------------------------------------------------------------------------
// ctx = k^T v per (b,h):  [32x4096] x [4096x32] -> ctxT[b][h][e][d] f16.
// One wave per (b,h): 2x2 wmma tiles, K-loop over 4096.
// ---------------------------------------------------------------------------
__global__ void te_ctx(const _Float16* __restrict__ k16,
                       const _Float16* __restrict__ v16m,
                       _Float16* __restrict__ ctxT) {
  int lane = threadIdx.x & 31;
  int h    = threadIdx.x >> 5;
  int b    = blockIdx.x;
  const _Float16* kb_ = k16  + (long)b * SEQ * DMODEL + h * DHEAD;
  const _Float16* vb_ = v16m + (long)b * SEQ * DMODEL + h * DHEAD;
  v8f c00 = {}, c01 = {}, c10 = {}, c11 = {};
  for (int k0 = 0; k0 < SEQ; k0 += 32) {
    v16h a0 = load_a_colmajor(kb_, 0,  k0, lane);
    v16h a1 = load_a_colmajor(kb_, 16, k0, lane);
    v16h b0 = load_b_rowmajor(vb_, 0,  k0, lane);
    v16h b1 = load_b_rowmajor(vb_, 16, k0, lane);
    c00 = wmma_f16(a0, b0, c00);
    c01 = wmma_f16(a0, b1, c01);
    c10 = wmma_f16(a1, b0, c10);
    c11 = wmma_f16(a1, b1, c11);
  }
  _Float16* ct = ctxT + ((long)b * NHEAD + h) * DHEAD * DHEAD;  // [e][d]
  int rb = (lane >> 4) << 3;
  int nc = lane & 15;
#pragma unroll
  for (int r = 0; r < 8; ++r) {
    ct[(long)(nc)      * DHEAD + rb + r]      = (_Float16)c00[r];
    ct[(long)(nc + 16) * DHEAD + rb + r]      = (_Float16)c01[r];
    ct[(long)(nc)      * DHEAD + 16 + rb + r] = (_Float16)c10[r];
    ct[(long)(nc + 16) * DHEAD + 16 + rb + r] = (_Float16)c11[r];
  }
}

// ---------------------------------------------------------------------------
// attn = q @ ctx per (b,h): 16x32 tile per wave, single K=32 wmma step.
// ---------------------------------------------------------------------------
__global__ void te_attn(const _Float16* __restrict__ q16,
                        const _Float16* __restrict__ ctxT,
                        _Float16* __restrict__ attn16) {
  int lane = threadIdx.x & 31;
  int wv   = threadIdx.x >> 5;
  long w   = (long)blockIdx.x * 8 + wv;
  int mt   = (int)(w >> 3);
  int h    = (int)(w & 7);
  int m0   = mt << 4;
  int b    = m0 >> 12;
  const _Float16* ct = ctxT + ((long)b * NHEAD + h) * DHEAD * DHEAD;
  v16h a  = load_a_frag(q16 + h * DHEAD, DMODEL, m0, 0, lane);
  v16h b0 = load_b_frag(ct, DHEAD, 0,  0, lane);
  v16h b1 = load_b_frag(ct, DHEAD, 16, 0, lane);
  v8f c0 = {}, c1 = {};
  c0 = wmma_f16(a, b0, c0);
  c1 = wmma_f16(a, b1, c1);
  int rb = (lane >> 4) << 3;
  int nc = lane & 15;
#pragma unroll
  for (int r = 0; r < 8; ++r) {
    long row = m0 + rb + r;
    attn16[row * DMODEL + h * DHEAD + nc]      = (_Float16)c0[r];
    attn16[row * DMODEL + h * DHEAD + 16 + nc] = (_Float16)c1[r];
  }
}

// ---------------------------------------------------------------------------
// Weight convert+transpose: W[K][N] f32 -> Wt[N][K] f16
// ---------------------------------------------------------------------------
__global__ void te_convw(const float* __restrict__ W, _Float16* __restrict__ Wt,
                         int K, int N) {
  long idx = (long)blockIdx.x * 256 + threadIdx.x;
  if (idx >= (long)K * N) return;
  int k = (int)(idx / N), n = (int)(idx % N);
  Wt[(long)n * K + k] = (_Float16)W[idx];
}

// ---------------------------------------------------------------------------
extern "C" void kernel_launch(void* const* d_in, const int* in_sizes, int n_in,
                              void* d_out, int out_size, void* d_ws,
                              size_t ws_size, hipStream_t stream) {
  (void)in_sizes; (void)n_in; (void)out_size; (void)ws_size;
  const float* x_in = (const float*)d_in[0];
  const float* Wq  = (const float*)d_in[1];
  const float* Wk  = (const float*)d_in[2];
  const float* Wv  = (const float*)d_in[3];
  const float* Wo  = (const float*)d_in[4];
  const float* bo  = (const float*)d_in[5];
  const float* g1  = (const float*)d_in[6];
  const float* b1  = (const float*)d_in[7];
  const float* W1  = (const float*)d_in[8];
  const float* bf1 = (const float*)d_in[9];
  const float* W2  = (const float*)d_in[10];
  const float* bf2 = (const float*)d_in[11];
  const float* g2  = (const float*)d_in[12];
  const float* b2  = (const float*)d_in[13];

  const size_t MB = 1024UL * 1024UL;
  char* ws = (char*)d_ws;
  float*    x    = (float*)ws;                       //  64 MB residual stream
  _Float16* h16  = (_Float16*)(ws + 64 * MB);        //  32 MB (also attn16)
  _Float16* q16  = (_Float16*)(ws + 96 * MB);        //  32 MB
  _Float16* k16  = (_Float16*)(ws + 128 * MB);       //  32 MB
  _Float16* v16m = (_Float16*)(ws + 160 * MB);       //  32 MB
  float*    q32  = (float*)(ws + 192 * MB);          //  64 MB \ alias ffh
  float*    k32  = (float*)(ws + 256 * MB);          //  64 MB /
  _Float16* ffh  = (_Float16*)(ws + 192 * MB);       // 128 MB (FF phase only)
  _Float16* ctxT = (_Float16*)(ws + 320 * MB);       // 256 KB
  _Float16* wts  = (_Float16*)(ws + 321 * MB);       //   6 MB f16 weights

  _Float16* WqT = wts;
  _Float16* WkT = WqT + (long)NLAYER * 65536;
  _Float16* WvT = WkT + (long)NLAYER * 65536;
  _Float16* WoT = WvT + (long)NLAYER * 65536;
  _Float16* W1T = WoT + (long)NLAYER * 65536;    // [1024][256] per layer
  _Float16* W2T = W1T + (long)NLAYER * 262144;   // [256][1024] per layer

  // Convert all weights up front (f32 -> transposed f16).
  for (int l = 0; l < NLAYER; ++l) {
    te_convw<<<65536 / 256, 256, 0, stream>>>(Wq + (long)l * 65536, WqT + (long)l * 65536, 256, 256);
    te_convw<<<65536 / 256, 256, 0, stream>>>(Wk + (long)l * 65536, WkT + (long)l * 65536, 256, 256);
    te_convw<<<65536 / 256, 256, 0, stream>>>(Wv + (long)l * 65536, WvT + (long)l * 65536, 256, 256);
    te_convw<<<65536 / 256, 256, 0, stream>>>(Wo + (long)l * 65536, WoT + (long)l * 65536, 256, 256);
    te_convw<<<262144 / 256, 256, 0, stream>>>(W1 + (long)l * 262144, W1T + (long)l * 262144, 256, 1024);
    te_convw<<<262144 / 256, 256, 0, stream>>>(W2 + (long)l * 262144, W2T + (long)l * 262144, 1024, 256);
  }

  // x = input + positional encoding
  te_pe_add<<<(int)(NROW * DMODEL / 256), 256, 0, stream>>>(x_in, x);

  auto gemm_blocks = [](int M, int N) { return (M / 128) * (N / 64); };

  for (int l = 0; l < NLAYER; ++l) {
    // ---- attention ----
    te_layernorm<<<(int)(NROW / 8), 256, 0, stream>>>(x, g1 + l * 256, b1 + l * 256, h16);
    te_gemm<EP_F32, 256><<<gemm_blocks(65536, 256), 256, 0, stream>>>(
        h16, WqT + (long)l * 65536, nullptr, nullptr, q32, 65536, 256);
    te_gemm<EP_F32, 256><<<gemm_blocks(65536, 256), 256, 0, stream>>>(
        h16, WkT + (long)l * 65536, nullptr, nullptr, k32, 65536, 256);
    te_gemm<EP_F16, 256><<<gemm_blocks(65536, 256), 256, 0, stream>>>(
        h16, WvT + (long)l * 65536, nullptr, nullptr, v16m, 65536, 256);
    te_qsoftmax<<<(int)(NROW * NHEAD / 8), 256, 0, stream>>>(q32, q16);
    te_ksoftmax<<<BATCH * DMODEL, 256, 0, stream>>>(k32, k16);
    te_ctx<<<BATCH, 256, 0, stream>>>(k16, v16m, ctxT);
    te_attn<<<(int)(NROW / 16 * NHEAD / 8), 256, 0, stream>>>(q16, ctxT, h16);
    te_gemm<EP_BIAS_RES, 256><<<gemm_blocks(65536, 256), 256, 0, stream>>>(
        h16, WoT + (long)l * 65536, bo + l * 256, x, x, 65536, 256);
    // ---- feedforward ----
    te_layernorm<<<(int)(NROW / 8), 256, 0, stream>>>(x, g2 + l * 256, b2 + l * 256, h16);
    te_gemm<EP_BIAS_GELU, 256><<<gemm_blocks(65536, 1024), 256, 0, stream>>>(
        h16, W1T + (long)l * 262144, bf1 + l * 1024, nullptr, ffh, 65536, 1024);
    te_gemm<EP_BIAS_RES, 1024><<<gemm_blocks(65536, 256), 256, 0, stream>>>(
        ffh, W2T + (long)l * 262144, bf2 + l * 256, x, x, 65536, 256);
  }

  hipMemcpyAsync(d_out, x, NROW * DMODEL * sizeof(float),
                 hipMemcpyDeviceToDevice, stream);
}
